// BilinearCrossAttnOnlyBlock_3264175145729
// MI455X (gfx1250) — matrix operand order repaired and checked by reading, not compiled, not run on hardware
//
#include <hip/hip_runtime.h>
#include <hip/hip_bf16.h>

typedef __attribute__((ext_vector_type(16))) _Float16 v16h;
typedef __attribute__((ext_vector_type(8)))  _Float16 v8h;
typedef __attribute__((ext_vector_type(8)))  float    v8f;

#define Bc  2
#define Lc  2048
#define Dc  1024
#define Hc  16
#define HDc 64
#define Rc  16

union F16x16 { v16h v; v8h h[2]; };

// WMMA D = A(16x32 f16) * B(32x16 f16) + C(16x16 f32)
__device__ __forceinline__ v8f wmma16(v16h a, v16h b, v8f c) {
  return __builtin_amdgcn_wmma_f32_16x16x32_f16(false, a, false, b, (short)0, c, false, false);
}

// Async DMA: 16 bytes global -> LDS, tracked by ASYNCcnt (CDNA5 GLOBAL_LOAD_ASYNC_TO_LDS_B128).
// lds generic pointers carry the LDS byte offset in their low 32 bits (aperture layout).
__device__ __forceinline__ void async_a128(const _Float16* g, _Float16* l) {
  unsigned loff = (unsigned)(uintptr_t)l;
  unsigned long long ga = (unsigned long long)(uintptr_t)g;
  asm volatile("global_load_async_to_lds_b128 %0, %1, off"
               :: "v"(loff), "v"(ga) : "memory");
}
__device__ __forceinline__ void wait_async0() {
  asm volatile("s_wait_asynccnt 0x0" ::: "memory");
}

// A fragment, 16x32 f16 row-major with 'stride' halfs/row.
// ISA layout: lanes 0-15 row=lane, halfs0-7 = K0-7, halfs8-15 = K16-23;
//             lanes16-31 row=lane-16, halfs0-7 = K8-15, halfs8-15 = K24-31.
__device__ __forceinline__ v16h ld_fragA(const _Float16* base, int stride) {
  int lane = threadIdx.x & 31;
  int r = lane & 15;
  int koff = (lane < 16) ? 0 : 8;
  F16x16 f;
  f.h[0] = *(const v8h*)(base + (size_t)r * stride + koff);
  f.h[1] = *(const v8h*)(base + (size_t)r * stride + koff + 16);
  return f.v;
}

// B fragment from N-major storage (row = N column, contiguous K halfs).
// ISA layout: lanes 0-15 col=lane, halfs0-7 = K0-7, halfs8-15 = K16-23; lanes16-31 col=lane-16, halfs0-7 = K8-15, halfs8-15 = K24-31.
__device__ __forceinline__ v16h ld_fragB(const _Float16* base, int stride) {
  int lane = threadIdx.x & 31;
  int n = lane & 15;
  int koff = (lane < 16) ? 0 : 8;
  F16x16 f;
  f.h[0] = *(const v8h*)(base + (size_t)n * stride + koff);
  f.h[1] = *(const v8h*)(base + (size_t)n * stride + koff + 16);
  return f.v;
}

// ---------------- elementwise helpers ----------------

__global__ __launch_bounds__(256) void cast_f32_f16(const float* __restrict__ in,
                                                    _Float16* __restrict__ out, int n) {
  int i = blockIdx.x * 256 + threadIdx.x;
  if (i < n) out[i] = (_Float16)in[i];
}

// U[H,HD,R] -> Ut[H,R,HD] fp16 (scale folded in)
__global__ __launch_bounds__(256) void uvt_kernel(const float* __restrict__ U,
                                                  _Float16* __restrict__ Ut, float scale) {
  int i = blockIdx.x * 256 + threadIdx.x;
  if (i >= Hc * HDc * Rc) return;
  int r  = i % Rc;
  int hd = (i / Rc) % HDc;
  int h  = i / (Rc * HDc);
  Ut[((size_t)h * Rc + r) * HDc + hd] = (_Float16)(U[i] * scale);
}

// One block per row: LayerNorm(x) * g + b -> fp16
__global__ __launch_bounds__(256) void layernorm_f16(const float* __restrict__ x,
                                                     const float* __restrict__ g,
                                                     const float* __restrict__ be,
                                                     _Float16* __restrict__ out) {
  __shared__ float sw[8], ssw[8];
  int row = blockIdx.x;
  int tid = threadIdx.x, lane = tid & 31, wave = tid >> 5;
  const float* xr = x + (size_t)row * Dc;
  float s = 0.f, ss = 0.f;
  for (int j = tid; j < Dc; j += 256) { float v = xr[j]; s += v; ss += v * v; }
  #pragma unroll
  for (int d = 1; d < 32; d <<= 1) { s += __shfl_xor(s, d, 32); ss += __shfl_xor(ss, d, 32); }
  if (lane == 0) { sw[wave] = s; ssw[wave] = ss; }
  __syncthreads();
  s = 0.f; ss = 0.f;
  #pragma unroll
  for (int i = 0; i < 8; ++i) { s += sw[i]; ss += ssw[i]; }
  float mean = s / Dc;
  float var  = ss / Dc - mean * mean;
  float rstd = rsqrtf(var + 1e-5f);
  _Float16* orow = out + (size_t)row * Dc;
  for (int j = tid; j < Dc; j += 256)
    orow[j] = (_Float16)((xr[j] - mean) * rstd * g[j] + be[j]);
}

// ---------------- tiled WMMA GEMM ----------------
// C[M,N] = A[M,K](f16) @ B[K,N](f16) + bias
// MODE 0: store fp16 [M,N]
// MODE 1: store fp16 transposed per head: vt[b,h,hd,L]  (M = B*L, N = D)
// MODE 2: store fp32 [M,N] = acc + bias + resid, plus fp16 mirror
template <int MODE>
__global__ __launch_bounds__(256)
void gemm_wmma(const _Float16* __restrict__ A, const _Float16* __restrict__ Bw,
               const float* __restrict__ bias, const float* __restrict__ resid,
               _Float16* __restrict__ outh, float* __restrict__ outf,
               int M, int N, int K) {
  __shared__ _Float16 a_s[64 * 40];    // 64x32 K-tile, padded rows
  __shared__ _Float16 b_s[128 * 40];   // N-major (transposed) 128x32
  const int tid = threadIdx.x, lane = tid & 31, wave = tid >> 5;
  const int nblk = N / 128;
  const int m0 = (blockIdx.x / nblk) * 64;
  const int n0 = (blockIdx.x % nblk) * 128;
  const int wm0 = (wave & 1) * 32;
  const int wn0 = (wave >> 1) * 32;
  v8f z = {0.f,0.f,0.f,0.f,0.f,0.f,0.f,0.f};
  v8f acc[2][2] = {{z, z}, {z, z}};

  const int arow = tid >> 2, akc = (tid & 3) * 8;
  const int bk = tid >> 4, bn = (tid & 15) * 8;

  for (int k0 = 0; k0 < K; k0 += 32) {
    // A tile: async DMA straight into LDS (no VGPR round-trip), 16B per thread
    async_a128(&A[(size_t)(m0 + arow) * K + k0 + akc], &a_s[arow * 40 + akc]);
    // L2 prefetch of next iteration's tiles while this one computes
    if (k0 + 32 < K) {
      __builtin_prefetch(&A[(size_t)(m0 + arow) * K + k0 + 32 + akc], 0, 3);
      __builtin_prefetch(&Bw[(size_t)(k0 + 32 + bk) * N + n0 + bn], 0, 3);
    }
    // B tile: coalesced global read, transposed scatter into LDS (N-major)
    #pragma unroll
    for (int pass = 0; pass < 2; ++pass) {
      int k = bk + pass * 16;
      v8h bv = *(const v8h*)&Bw[(size_t)(k0 + k) * N + n0 + bn];
      #pragma unroll
      for (int j = 0; j < 8; ++j) b_s[(bn + j) * 40 + k] = bv[j];
    }
    wait_async0();
    __syncthreads();
    v16h af0 = ld_fragA(&a_s[(wm0) * 40], 40);
    v16h af1 = ld_fragA(&a_s[(wm0 + 16) * 40], 40);
    v16h bf0 = ld_fragB(&b_s[(wn0) * 40], 40);
    v16h bf1 = ld_fragB(&b_s[(wn0 + 16) * 40], 40);
    acc[0][0] = wmma16(af0, bf0, acc[0][0]);
    acc[0][1] = wmma16(af0, bf1, acc[0][1]);
    acc[1][0] = wmma16(af1, bf0, acc[1][0]);
    acc[1][1] = wmma16(af1, bf1, acc[1][1]);
    __syncthreads();
  }

  const int rowoff = (lane < 16) ? 0 : 8;
  #pragma unroll
  for (int tm = 0; tm < 2; ++tm) {
    #pragma unroll
    for (int tn = 0; tn < 2; ++tn) {
      int nG = n0 + wn0 + tn * 16 + (lane & 15);
      int r0 = m0 + wm0 + tm * 16 + rowoff;
      float bval = bias[nG];
      if (MODE == 0) {
        #pragma unroll
        for (int i = 0; i < 8; ++i)
          outh[(size_t)(r0 + i) * N + nG] = (_Float16)(acc[tm][tn][i] + bval);
      } else if (MODE == 1) {
        int hh = nG >> 6, hd = nG & 63;
        int bb = r0 >> 11;        // / Lc
        int key0 = r0 & (Lc - 1); // r0..r0+7 stay in one batch (aligned 8)
        v8h pk;
        #pragma unroll
        for (int i = 0; i < 8; ++i) pk[i] = (_Float16)(acc[tm][tn][i] + bval);
        *(v8h*)&outh[(((size_t)bb * Hc + hh) * HDc + hd) * Lc + key0] = pk;
      } else {
        #pragma unroll
        for (int i = 0; i < 8; ++i) {
          size_t o = (size_t)(r0 + i) * N + nG;
          float v = acc[tm][tn][i] + bval + resid[o];
          outf[o] = v;
          outh[o] = (_Float16)v;
        }
      }
    }
  }
}

// ---------------- low-rank projection: out[b,h,l,0:16] = X[b,l,h,:] @ Wt[h]^T ----------------
// Wt is [H,R,HD] (N-major over R, contiguous over K=HD). Output [B,H,L,32], cols 16..31 zero.
__global__ __launch_bounds__(256)
void lowrank_kernel(const _Float16* __restrict__ Xh, const _Float16* __restrict__ Wt,
                    _Float16* __restrict__ out) {
  int w = (blockIdx.x * 256 + threadIdx.x) >> 5;
  int lane = threadIdx.x & 31;
  const int L16 = Lc / 16;
  int b  = w / (Hc * L16);
  int h  = (w / L16) % Hc;
  int m0 = (w % L16) * 16;
  const _Float16* abase = Xh + ((size_t)b * Lc + m0) * Dc + h * HDc;
  const _Float16* bbase = Wt + (size_t)h * Rc * HDc;
  v8f acc = {0.f,0.f,0.f,0.f,0.f,0.f,0.f,0.f};
  acc = wmma16(ld_fragA(abase, Dc),      ld_fragB(bbase, HDc),      acc);
  acc = wmma16(ld_fragA(abase + 32, Dc), ld_fragB(bbase + 32, HDc), acc);
  int r = lane & 15, rowoff = (lane < 16) ? 0 : 8;
  #pragma unroll
  for (int i = 0; i < 8; ++i) {
    size_t o = (((size_t)b * Hc + h) * Lc + m0 + rowoff + i) * 32;
    out[o + r]      = (_Float16)acc[i];
    out[o + 16 + r] = (_Float16)0.0f;
  }
}

// ---------------- flash attention over rank-16 scores ----------------
// One wave per 16-query tile of one (b,h). ql/kl: [B,H,L,32] f16, vt: [B,H,HD,L] f16.
__global__ __launch_bounds__(256)
void attn_kernel(const _Float16* __restrict__ ql, const _Float16* __restrict__ kl,
                 const _Float16* __restrict__ vt, const unsigned char* __restrict__ mask,
                 _Float16* __restrict__ attn) {
  __shared__ _Float16 p_s[8 * 16 * 32];   // per-wave P staging (C-layout -> A-layout)
  int tid = threadIdx.x, lane = tid & 31, wib = tid >> 5;
  int w = (blockIdx.x * 256 + tid) >> 5;
  const int L16 = Lc / 16;
  int b  = w / (Hc * L16);
  int h  = (w / L16) % Hc;
  int m0 = (w % L16) * 16;

  const _Float16* qlp = ql + (((size_t)b * Hc + h) * Lc + m0) * 32;
  const _Float16* klp = kl + (((size_t)b * Hc + h) * Lc) * 32;
  const _Float16* vtp = vt + ((size_t)b * Hc + h) * HDc * Lc;
  const unsigned char* mrow = mask + (size_t)b * Lc;
  _Float16* pw = &p_s[wib * 16 * 32];

  v16h qfrag = ld_fragA(qlp, 32);
  v8f z = {0.f,0.f,0.f,0.f,0.f,0.f,0.f,0.f};
  v8f O[4] = {z, z, z, z};
  float rm[8], rs[8];
  #pragma unroll
  for (int i = 0; i < 8; ++i) { rm[i] = -1e30f; rs[i] = 0.f; }
  const int n = lane & 15;
  const int rowoff = (lane < 16) ? 0 : 8;

  for (int kt = 0; kt < Lc; kt += 32) {
    v16h kf0 = ld_fragB(klp + (size_t)kt * 32, 32);
    v16h kf1 = ld_fragB(klp + (size_t)(kt + 16) * 32, 32);
    v8f S0 = wmma16(qfrag, kf0, z);
    v8f S1 = wmma16(qfrag, kf1, z);
    // padding mask: lane's column = one key index, applies to all 8 rows
    float m0v = mrow[kt + n]      ? -1e9f : 0.f;
    float m1v = mrow[kt + 16 + n] ? -1e9f : 0.f;
    #pragma unroll
    for (int i = 0; i < 8; ++i) {
      if (m0v != 0.f) S0[i] = m0v;
      if (m1v != 0.f) S1[i] = m1v;
    }
    float corr[8];
    #pragma unroll
    for (int i = 0; i < 8; ++i) {
      float mx = fmaxf(S0[i], S1[i]);
      #pragma unroll
      for (int d = 1; d < 16; d <<= 1) mx = fmaxf(mx, __shfl_xor(mx, d, 32));
      float nm = fmaxf(rm[i], mx);
      corr[i] = __expf(rm[i] - nm);
      rm[i] = nm;
      float p0 = __expf(S0[i] - nm);
      float p1 = __expf(S1[i] - nm);
      float ps = p0 + p1;
      #pragma unroll
      for (int d = 1; d < 16; d <<= 1) ps += __shfl_xor(ps, d, 32);
      rs[i] = rs[i] * corr[i] + ps;
      int row = i + rowoff;
      pw[row * 32 + n]      = (_Float16)p0;
      pw[row * 32 + 16 + n] = (_Float16)p1;
    }
    #pragma unroll
    for (int c = 0; c < 4; ++c) {
      #pragma unroll
      for (int i = 0; i < 8; ++i) O[c][i] *= corr[i];
    }
    asm volatile("s_wait_dscnt 0x0" ::: "memory");   // same-wave LDS RAW fence
    v16h pf = ld_fragA(pw, 32);
    #pragma unroll
    for (int c = 0; c < 4; ++c) {
      v16h vf = ld_fragB(vtp + (size_t)(c * 16) * Lc + kt, Lc);
      O[c] = wmma16(pf, vf, O[c]);
    }
  }

  #pragma unroll
  for (int c = 0; c < 4; ++c) {
    #pragma unroll
    for (int i = 0; i < 8; ++i) {
      int row = m0 + i + rowoff;
      attn[((size_t)b * Lc + row) * Dc + h * HDc + c * 16 + n] =
          (_Float16)(O[c][i] / rs[i]);
    }
  }
}

// ---------------- host launch ----------------
extern "C" void kernel_launch(void* const* d_in, const int* in_sizes, int n_in,
                              void* d_out, int out_size, void* d_ws, size_t ws_size,
                              hipStream_t stream) {
  const float* x_seq    = (const float*)d_in[0];
  const float* x_struct = (const float*)d_in[1];
  const unsigned char* pmask = (const unsigned char*)d_in[2];

  const size_t BLD = (size_t)Bc * Lc * Dc;           // 4,194,304
  char* w = (char*)d_ws;
  size_t off = 0;
  auto alloc = [&](size_t bytes) { off = (off + 255) & ~(size_t)255; size_t o = off; off += bytes; return o; };

  _Float16* q_in_h = (_Float16*)(w + alloc(BLD * 2));
  _Float16* kv_h   = (_Float16*)(w + alloc(BLD * 2));
  _Float16* q_h    = (_Float16*)(w + alloc(BLD * 2));
  _Float16* k_h    = (_Float16*)(w + alloc(BLD * 2));
  _Float16* vt_h   = (_Float16*)(w + alloc(BLD * 2));
  _Float16* attn_h = (_Float16*)(w + alloc(BLD * 2));
  _Float16* kv2_h  = (_Float16*)(w + alloc(BLD * 2));
  _Float16* Wq_h   = (_Float16*)(w + alloc((size_t)Dc * Dc * 2));
  _Float16* Wk_h   = (_Float16*)(w + alloc((size_t)Dc * Dc * 2));
  _Float16* Wv_h   = (_Float16*)(w + alloc((size_t)Dc * Dc * 2));
  _Float16* Wo_h   = (_Float16*)(w + alloc((size_t)Dc * Dc * 2));
  _Float16* Ut_h   = (_Float16*)(w + alloc((size_t)Hc * Rc * HDc * 2));
  _Float16* Vt_h   = (_Float16*)(w + alloc((size_t)Hc * Rc * HDc * 2));
  _Float16* ql_h   = (_Float16*)(w + alloc((size_t)Bc * Hc * Lc * 32 * 2));
  _Float16* kl_h   = (_Float16*)(w + alloc((size_t)Bc * Hc * Lc * 32 * 2));

  const int DD_BLOCKS  = (Dc * Dc) / 256;            // 4096
  const int BLD_BLOCKS = (int)(BLD / 256);           // 16384
  const int GEMM_BLOCKS = (Bc * Lc / 64) * (Dc / 128); // 512
  const int WAVE_BLOCKS = (Bc * Hc * (Lc / 16)) / 8;   // 512

  for (int p = 0; p < 2; ++p) {
    const float* xq  = (p == 0) ? x_seq : x_struct;
    const float* g   = (const float*)d_in[3 + p * 2];
    const float* be  = (const float*)d_in[4 + p * 2];
    int base = 7 + p * 10;
    const float* Wq = (const float*)d_in[base + 0];
    const float* bq = (const float*)d_in[base + 1];
    const float* Wk = (const float*)d_in[base + 2];
    const float* bk = (const float*)d_in[base + 3];
    const float* Wv = (const float*)d_in[base + 4];
    const float* bv = (const float*)d_in[base + 5];
    const float* U  = (const float*)d_in[base + 6];
    const float* V  = (const float*)d_in[base + 7];
    const float* Wo = (const float*)d_in[base + 8];
    const float* bo = (const float*)d_in[base + 9];
    float* outp = (float*)d_out + (size_t)p * BLD;
    const _Float16* kvp = (p == 0) ? kv_h : kv2_h;   // pass 2 KV = updated x_seq (fp16 mirror)

    cast_f32_f16<<<DD_BLOCKS, 256, 0, stream>>>(Wq, Wq_h, Dc * Dc);
    cast_f32_f16<<<DD_BLOCKS, 256, 0, stream>>>(Wk, Wk_h, Dc * Dc);
    cast_f32_f16<<<DD_BLOCKS, 256, 0, stream>>>(Wv, Wv_h, Dc * Dc);
    cast_f32_f16<<<DD_BLOCKS, 256, 0, stream>>>(Wo, Wo_h, Dc * Dc);
    uvt_kernel<<<64, 256, 0, stream>>>(U, Ut_h, 0.25f);  // 1/sqrt(R) folded into ql
    uvt_kernel<<<64, 256, 0, stream>>>(V, Vt_h, 1.0f);
    if (p == 0)
      cast_f32_f16<<<BLD_BLOCKS, 256, 0, stream>>>(x_struct, kv_h, (int)BLD);
    layernorm_f16<<<Bc * Lc, 256, 0, stream>>>(xq, g, be, q_in_h);

    gemm_wmma<0><<<GEMM_BLOCKS, 256, 0, stream>>>(q_in_h, Wq_h, bq, nullptr, q_h, nullptr,
                                                  Bc * Lc, Dc, Dc);
    gemm_wmma<0><<<GEMM_BLOCKS, 256, 0, stream>>>(kvp, Wk_h, bk, nullptr, k_h, nullptr,
                                                  Bc * Lc, Dc, Dc);
    gemm_wmma<1><<<GEMM_BLOCKS, 256, 0, stream>>>(kvp, Wv_h, bv, nullptr, vt_h, nullptr,
                                                  Bc * Lc, Dc, Dc);

    lowrank_kernel<<<WAVE_BLOCKS, 256, 0, stream>>>(q_h, Ut_h, ql_h);
    lowrank_kernel<<<WAVE_BLOCKS, 256, 0, stream>>>(k_h, Vt_h, kl_h);

    attn_kernel<<<WAVE_BLOCKS, 256, 0, stream>>>(ql_h, kl_h, vt_h, pmask, attn_h);

    gemm_wmma<2><<<GEMM_BLOCKS, 256, 0, stream>>>(attn_h, Wo_h, bo, xq, kv2_h, outp,
                                                  Bc * Lc, Dc, Dc);
  }
}
